// MoFE_48017734369472
// MI455X (gfx1250) — matrix operand order repaired and checked by reading, not compiled, not run on hardware
//
#include <hip/hip_runtime.h>
#include <hip/hip_bf16.h>
#include <math.h>

// ---------------------------------------------------------------------------
// MoFE on gfx1250 (MI455X):
//   K1: global max+mean pool over HxW  -> pooled (B=8, C=96) in ws
//   K2: one-wave gate: 3x v_wmma_f32_16x16x32_f16 over K=96 (both FCs fused
//       into N=12), branch-free b128 LDS fragment loads, fp32 epilogue
//   K3: fused dwconv3x3 -> ReLU -> dwconv3x3, gate-weighted accumulation,
//       32x32 tiles; tile staged with global_load_async_to_lds_b32 (ASYNCcnt)
// ---------------------------------------------------------------------------

#define DIM 96
#define NE  6
#define BATCH 8
#define HW 192
#define PLANE (HW * HW)          // 36864

typedef __attribute__((ext_vector_type(16))) _Float16 v16h;
typedef __attribute__((ext_vector_type(8)))  _Float16 v8h;
typedef __attribute__((ext_vector_type(8)))  float    v8f;

// ---------------- Kernel 1: pooled[b][c] = max(x[b,c]) + mean(x[b,c]) -------
__global__ __launch_bounds__(256) void mofe_pool_kernel(
    const float* __restrict__ x, float* __restrict__ pooled) {
  __shared__ float smax[256];
  __shared__ float ssum[256];
  const int tid = threadIdx.x;
  const float4* xv =
      (const float4*)(x + (size_t)blockIdx.x * (size_t)PLANE);
  float m = -INFINITY, s = 0.0f;
  for (int i = tid; i < PLANE / 4; i += 256) {
    float4 v = xv[i];
    m = fmaxf(m, fmaxf(fmaxf(v.x, v.y), fmaxf(v.z, v.w)));
    s += (v.x + v.y) + (v.z + v.w);
  }
  smax[tid] = m;
  ssum[tid] = s;
  __syncthreads();
  for (int off = 128; off > 0; off >>= 1) {
    if (tid < off) {
      smax[tid] = fmaxf(smax[tid], smax[tid + off]);
      ssum[tid] += ssum[tid + off];
    }
    __syncthreads();
  }
  if (tid == 0)
    pooled[blockIdx.x] = smax[0] + ssum[0] * (1.0f / (float)PLANE);
}

// ---------------- Kernel 2: gate (single wave32, uses v_wmma) ---------------
// S(16x16) = A(16x96) * B(96x16):
//   A rows 0..7  = pooled[b][k]                 (rows 8..15 zero-padded)
//   B cols 0..5  = w_fc1[n][k], 6..11 = w_fc0   (cols 12..15 zero-padded)
__global__ __launch_bounds__(32) void mofe_gate_kernel(
    const float* __restrict__ pooled,
    const float* __restrict__ w_fc0, const float* __restrict__ b_fc0,
    const float* __restrict__ w_fc1, const float* __restrict__ b_fc1,
    float* __restrict__ cof) {
  __shared__ _Float16 Ph[16][DIM];    // padded pooled (f16)
  __shared__ _Float16 Wh[16][DIM];    // padded combined fc weights (f16)
  __shared__ float S[16][16];         // wmma result

  const int lane = threadIdx.x;
  for (int i = lane; i < 16 * DIM; i += 32) {
    int r = i / DIM, k = i - r * DIM;
    Ph[r][k] = (r < BATCH) ? (_Float16)pooled[r * DIM + k] : (_Float16)0.0f;
    float wv = 0.0f;
    if (r < NE)       wv = w_fc1[r * DIM + k];
    else if (r < 12)  wv = w_fc0[(r - NE) * DIM + k];
    Wh[r][k] = (_Float16)wv;
  }
  __syncthreads();

  const int mn  = lane & 15;          // A row (M) / B col (N)
  const int hiA = (lane >> 4) * 8;    // A: upper half-wave carries K+8
  const int hiB = (lane >> 4) * 8;    // B: upper half-wave carries K+8

  v8f c = {};
#pragma unroll
  for (int chunk = 0; chunk < 3; ++chunk) {
    const int k0 = chunk * 32;
    v16h a, b;
    // A 16x32 f16 layout: halves h<8 -> K=k0+hiA+h; h>=8 -> K=k0+hiA+16+(h-8)
    ((v8h*)&a)[0] = *(const v8h*)&Ph[mn][k0 + hiA];
    ((v8h*)&a)[1] = *(const v8h*)&Ph[mn][k0 + hiA + 16];
    // B 32x16 f16 layout: halves h<8 -> K=k0+hiB+h; h>=8 -> K=k0+hiB+16+(h-8)
    ((v8h*)&b)[0] = *(const v8h*)&Wh[mn][k0 + hiB];
    ((v8h*)&b)[1] = *(const v8h*)&Wh[mn][k0 + hiB + 16];
    c = __builtin_amdgcn_wmma_f32_16x16x32_f16(
        /*neg_a=*/false, a, /*neg_b=*/false, b,
        /*c_mod=*/(short)0, c, /*reuse_a=*/false, /*reuse_b=*/false);
  }

  // C/D layout: VGPR r -> M = r + 8*(lane>=16), N = lane&15
#pragma unroll
  for (int r = 0; r < 8; ++r)
    S[r + ((lane >> 4) << 3)][mn] = c[r];
  __syncthreads();

  if (lane < BATCH) {
    const int b = lane;
    float g[NE], nz[NE];
#pragma unroll
    for (int e = 0; e < NE; ++e) {
      float r1 = S[b][e] + b_fc1[e];
      g[e] = (r1 > 0.0f) ? r1 : 0.2f * r1;                 // leaky_relu 0.2
      float r0 = S[b][6 + e] + b_fc0[e];
      nz[e] = (r0 > 20.0f) ? r0 : log1pf(expf(r0));        // softplus
    }
    float mu = 0.0f;
#pragma unroll
    for (int e = 0; e < NE; ++e) mu += nz[e];
    mu *= (1.0f / NE);
    float var = 0.0f;
#pragma unroll
    for (int e = 0; e < NE; ++e) { float d = nz[e] - mu; var += d * d; }
    float sd = sqrtf(var * (1.0f / (NE - 1)));             // ddof=1
    float sc[NE];
#pragma unroll
    for (int e = 0; e < NE; ++e) sc[e] = g[e] + (nz[e] - mu) / sd;

    bool sel[NE] = {false, false, false, false, false, false};
    for (int t = 0; t < 3; ++t) {                          // top-3 of scores
      int best = -1; float bv = -INFINITY;
#pragma unroll
      for (int e = 0; e < NE; ++e)
        if (!sel[e] && sc[e] > bv) { bv = sc[e]; best = e; }
      sel[best] = true;
    }
    float mg = -INFINITY;
#pragma unroll
    for (int e = 0; e < NE; ++e) if (sel[e]) mg = fmaxf(mg, g[e]);
    float ex[NE], den = 0.0f;
#pragma unroll
    for (int e = 0; e < NE; ++e) {
      ex[e] = sel[e] ? expf(g[e] - mg) : 0.0f;
      den += ex[e];
    }
    float inv = 1.0f / den;
#pragma unroll
    for (int e = 0; e < NE; ++e) cof[b * NE + e] = ex[e] * inv;
  }
}

// ---------------- Kernel 3: fused expert convs, 32x32 tiles ----------------
#define TS 32
#define XS_W 37   // 36 + pad
#define T1_W 35   // 34 + pad

// Async DMA one dword from global (base + byte_off) into LDS at lds_off.
// GVS form: mem_addr = SGPR_U64 + VGPR_I32; LDS dest addr in VDST VGPR.
__device__ __forceinline__ void async_ld_b32(unsigned lds_off,
                                             const float* __restrict__ base,
                                             unsigned byte_off) {
  asm volatile("global_load_async_to_lds_b32 %0, %1, %2"
               :: "v"(lds_off), "v"(byte_off), "s"(base)
               : "memory");
}

__device__ __forceinline__ void wait_asynccnt0() {
  asm volatile("s_wait_asynccnt 0" ::: "memory");
}

__global__ __launch_bounds__(256) void mofe_conv_kernel(
    const float* __restrict__ x,
    const float* __restrict__ ew1, const float* __restrict__ eb1,
    const float* __restrict__ ew2, const float* __restrict__ eb2,
    const float* __restrict__ cof, float* __restrict__ out) {
  __shared__ float xs[36][XS_W];   // input tile + halo 2
  __shared__ float t1[34][T1_W];   // relu(conv1) tile + halo 1

  const int tid = threadIdx.x;
  const int bc  = blockIdx.z;              // b*96 + c
  const int b   = bc / DIM;
  const int c   = bc - b * DIM;
  const int y0  = blockIdx.y * TS;
  const int x0  = blockIdx.x * TS;
  const float* __restrict__ xp = x + (size_t)bc * (size_t)PLANE;

  // stage x tile (36x36): async DMA in-bounds, ds_store zero at borders
  for (int i = tid; i < 36 * 36; i += 256) {
    int iy = i / 36, ix = i - iy * 36;
    int gy = y0 + iy - 2, gx = x0 + ix - 2;
    unsigned lds_off = (unsigned)(uintptr_t)&xs[iy][ix];
    if (gy >= 0 && gy < HW && gx >= 0 && gx < HW) {
      async_ld_b32(lds_off, xp, (unsigned)((gy * HW + gx) * 4));
    } else {
      xs[iy][ix] = 0.0f;
    }
  }
  wait_asynccnt0();
  __syncthreads();

  float acc0 = 0.0f, acc1 = 0.0f, acc2 = 0.0f, acc3 = 0.0f;

  for (int e = 0; e < NE; ++e) {
    const float w = cof[b * NE + e];       // uniform per block
    if (w == 0.0f) continue;               // skip non-top-k experts

    const float* __restrict__ k1p = ew1 + (size_t)(e * DIM + c) * 9;
    const float* __restrict__ k2p = ew2 + (size_t)(e * DIM + c) * 9;
    float k1[9], k2[9];
#pragma unroll
    for (int i = 0; i < 9; ++i) { k1[i] = k1p[i]; k2[i] = k2p[i]; }
    const float bb1 = eb1[e * DIM + c];
    const float bb2 = eb2[e * DIM + c];

    // conv1 + ReLU into t1 (34x34)
    for (int i = tid; i < 34 * 34; i += 256) {
      int iy = i / 34, ix = i - iy * 34;
      float a = bb1;
#pragma unroll
      for (int ky = 0; ky < 3; ++ky)
#pragma unroll
        for (int kx = 0; kx < 3; ++kx)
          a = fmaf(xs[iy + ky][ix + kx], k1[ky * 3 + kx], a);
      t1[iy][ix] = fmaxf(a, 0.0f);
    }
    __syncthreads();

    // conv2, weighted accumulate (each thread owns 4 output pixels)
#pragma unroll
    for (int j = 0; j < 4; ++j) {
      int i = tid + j * 256;
      int oy = i >> 5, ox = i & 31;
      float a = bb2;
#pragma unroll
      for (int ky = 0; ky < 3; ++ky)
#pragma unroll
        for (int kx = 0; kx < 3; ++kx)
          a = fmaf(t1[oy + ky][ox + kx], k2[ky * 3 + kx], a);
      float r = w * a;
      if (j == 0) acc0 += r; else if (j == 1) acc1 += r;
      else if (j == 2) acc2 += r; else acc3 += r;
    }
    __syncthreads();                       // t1 reused next expert
  }

  float* __restrict__ op = out + (size_t)bc * (size_t)PLANE;
#pragma unroll
  for (int j = 0; j < 4; ++j) {
    int i = tid + j * 256;
    int oy = i >> 5, ox = i & 31;
    float r = (j == 0) ? acc0 : (j == 1) ? acc1 : (j == 2) ? acc2 : acc3;
    op[(y0 + oy) * HW + (x0 + ox)] = r;
  }
}

// ---------------------------------------------------------------------------
extern "C" void kernel_launch(void* const* d_in, const int* in_sizes, int n_in,
                              void* d_out, int out_size, void* d_ws, size_t ws_size,
                              hipStream_t stream) {
  const float* x     = (const float*)d_in[0];
  const float* w_fc0 = (const float*)d_in[1];
  const float* b_fc0 = (const float*)d_in[2];
  const float* w_fc1 = (const float*)d_in[3];
  const float* b_fc1 = (const float*)d_in[4];
  const float* ew1   = (const float*)d_in[5];
  const float* eb1   = (const float*)d_in[6];
  const float* ew2   = (const float*)d_in[7];
  const float* eb2   = (const float*)d_in[8];
  float* out = (float*)d_out;

  float* ws     = (float*)d_ws;
  float* pooled = ws;                 // 768 floats
  float* cof    = ws + BATCH * DIM;   // 48 floats

  mofe_pool_kernel<<<BATCH * DIM, 256, 0, stream>>>(x, pooled);
  mofe_gate_kernel<<<1, 32, 0, stream>>>(pooled, w_fc0, b_fc0,
                                         w_fc1, b_fc1, cof);
  mofe_conv_kernel<<<dim3(HW / TS, HW / TS, BATCH * DIM), 256, 0, stream>>>(
      x, ew1, eb1, ew2, eb2, cof, out);
}
